// TraditionalMoEWithSkip_26199300505855
// MI455X (gfx1250) — matrix operand order repaired and checked by reading, not compiled, not run on hardware
//
#include <hip/hip_runtime.h>
#include <hip/hip_bf16.h>

typedef __attribute__((ext_vector_type(16))) _Float16 v16h;
typedef __attribute__((ext_vector_type(8)))  _Float16 v8h;
typedef __attribute__((ext_vector_type(8)))  float    v8f;

#define BN_EPS 1e-5f

// ---------------------------------------------------------------------------
// Weight repack: [4][64co][64ci][3][3] f32  ->  [4][9tap][64co][64ci] f16
// (B-matrix fragments become one contiguous 32B load per lane)
// ---------------------------------------------------------------------------
__global__ __launch_bounds__(256) void k_wcvt(const float* __restrict__ w,
                                              _Float16* __restrict__ wt) {
  int idx = blockIdx.x * 256 + threadIdx.x;           // 4*9*64*64 = 147456
  if (idx >= 4 * 9 * 64 * 64) return;
  int ci = idx & 63;
  int co = (idx >> 6) & 63;
  int t  = (idx >> 12) % 9;
  int e  = idx / (9 * 64 * 64);
  int ky = t / 3, kx = t % 3;
  wt[(((size_t)e * 9 + t) * 64 + co) * 64 + ci] =
      (_Float16)w[((((size_t)e * 64 + co) * 64 + ci) * 3 + ky) * 3 + kx];
}

// ---------------------------------------------------------------------------
// Expert stem conv: 3->64, 3x3 pad1 + BN + ReLU, x f32 NCHW -> act f16 NHWC
// (K=27 is too ragged for WMMA; negligible FLOPs -> naive)
// ---------------------------------------------------------------------------
__global__ __launch_bounds__(256) void k_econv1(const float* __restrict__ x,
    const float* __restrict__ w, const float* __restrict__ g,
    const float* __restrict__ b, const float* __restrict__ m,
    const float* __restrict__ v, _Float16* __restrict__ out) {
  int idx = blockIdx.x * 256 + threadIdx.x;           // 256*32*32*64
  int co = idx & 63, xx = (idx >> 6) & 31, yy = (idx >> 11) & 31, bb = idx >> 16;
  float acc = 0.f;
  for (int ci = 0; ci < 3; ++ci)
    for (int ky = 0; ky < 3; ++ky) {
      int iy = yy + ky - 1; if ((unsigned)iy >= 32u) continue;
      for (int kx = 0; kx < 3; ++kx) {
        int ix = xx + kx - 1; if ((unsigned)ix >= 32u) continue;
        acc += x[((bb * 3 + ci) * 32 + iy) * 32 + ix] *
               w[((co * 3 + ci) * 3 + ky) * 3 + kx];
      }
    }
  float sc = g[co] * rsqrtf(v[co] + BN_EPS);
  out[idx] = (_Float16)fmaxf((acc - m[co]) * sc + b[co], 0.f);
}

// ---------------------------------------------------------------------------
// Main compute: 64->64 3x3 pad1 conv as 9 tap-GEMMs on v_wmma_f32_16x16x32_f16
// in/out: f16 NHWC [256][32][32][64]; wt: f16 [9][co][ci]
// Fused BN (+ optional residual) + ReLU epilogue.
// Block = 1 image x 4-row band, 8 waves; wave = 16 positions x 64 channels.
// ---------------------------------------------------------------------------
__global__ __launch_bounds__(256) void k_conv64_wmma(
    const _Float16* __restrict__ in, const _Float16* __restrict__ wt,
    const float* __restrict__ bng, const float* __restrict__ bnb,
    const float* __restrict__ bnm, const float* __restrict__ bnv,
    const _Float16* __restrict__ res, _Float16* __restrict__ out) {
  __shared__ _Float16 sIn[6 * 34 * 64];               // 26112 B halo tile
  const int tid  = threadIdx.x;
  const int img  = blockIdx.x;
  const int band = blockIdx.y;                        // 0..7 (4 rows each)
  const int y0   = band * 4 - 1;

  // Stage input band (+halo) into LDS, zero-padded, 16B units
  for (int u = tid; u < 6 * 34 * 8; u += 256) {
    int c8 = u & 7, rx = (u >> 3) % 34, ry = u / (34 * 8);
    int gy = y0 + ry, gx = rx - 1;
    v8h val = {};
    if ((unsigned)gy < 32u && (unsigned)gx < 32u)
      val = *(const v8h*)(in + ((((size_t)img * 32 + gy) * 32 + gx) * 64 + c8 * 8));
    *(v8h*)(sIn + ((ry * 34 + rx) * 64 + c8 * 8)) = val;
  }
  __syncthreads();

  const int wv = tid >> 5, lane = tid & 31;
  const int hi = lane >> 4, l2 = lane & 15;
  const int row     = band * 4 + (wv >> 1);
  const int colbase = (wv & 1) * 16;
  const int lyb     = (wv >> 1) + 1;

  v8f acc0 = {}, acc1 = {}, acc2 = {}, acc3 = {};
  for (int t = 0; t < 9; ++t) {
    const int dy = t / 3 - 1, dx = t % 3 - 1;
    // A row m = l2 -> spatial col = colbase + l2 (shifted by tap)
    const _Float16* ap = sIn + ((lyb + dy) * 34 + (colbase + l2 + dx + 1)) * 64;
    const _Float16* wp = wt + (size_t)t * 64 * 64;
#pragma unroll
    for (int kb = 0; kb < 64; kb += 32) {
      // A 16x32 f16: lo-half lanes K=0-15, hi-half K=16-31 (contiguous)
      v8h a0 = *(const v8h*)(ap + kb + hi * 16);
      v8h a1 = *(const v8h*)(ap + kb + hi * 16 + 8);
      v16h a;
#pragma unroll
      for (int i = 0; i < 8; ++i) { a[i] = a0[i]; a[i + 8] = a1[i]; }
      // B 32x16 f16: col = l2, lo-half K=0-15, hi-half K=16-31 (contiguous ci)
      const _Float16* bp = wp + kb + hi * 16;
      v16h b0 = *(const v16h*)(bp + (size_t)(l2 +  0) * 64);
      v16h b1 = *(const v16h*)(bp + (size_t)(l2 + 16) * 64);
      v16h b2 = *(const v16h*)(bp + (size_t)(l2 + 32) * 64);
      v16h b3 = *(const v16h*)(bp + (size_t)(l2 + 48) * 64);
      acc0 = __builtin_amdgcn_wmma_f32_16x16x32_f16(false, a, false, b0, (short)0, acc0, false, false);
      acc1 = __builtin_amdgcn_wmma_f32_16x16x32_f16(false, a, false, b1, (short)0, acc1, false, false);
      acc2 = __builtin_amdgcn_wmma_f32_16x16x32_f16(false, a, false, b2, (short)0, acc2, false, false);
      acc3 = __builtin_amdgcn_wmma_f32_16x16x32_f16(false, a, false, b3, (short)0, acc3, false, false);
    }
  }

  // D layout: N = l2 (+16*nt), M = vgpr + 8*hi
#define CDNA5_EPILOG(ACC, NT) do {                                            \
    const int co = (NT) * 16 + l2;                                            \
    const float sc = bng[co] * rsqrtf(bnv[co] + BN_EPS);                      \
    const float sh = bnb[co] - bnm[co] * sc;                                  \
    _Pragma("unroll")                                                         \
    for (int vv = 0; vv < 8; ++vv) {                                          \
      const int col = colbase + vv + 8 * hi;                                  \
      const size_t oidx = (((size_t)img * 32 + row) * 32 + col) * 64 + co;    \
      float val = ACC[vv] * sc + sh;                                          \
      if (res) val += (float)res[oidx];                                       \
      out[oidx] = (_Float16)fmaxf(val, 0.f);                                  \
    }                                                                         \
  } while (0)
  CDNA5_EPILOG(acc0, 0);
  CDNA5_EPILOG(acc1, 1);
  CDNA5_EPILOG(acc2, 2);
  CDNA5_EPILOG(acc3, 3);
#undef CDNA5_EPILOG
}

// Expert feature: global average pool over 32x32 -> [256][64]
__global__ __launch_bounds__(256) void k_efeat(const _Float16* __restrict__ act,
                                               float* __restrict__ feat) {
  int idx = blockIdx.x * 256 + threadIdx.x;           // 256*64
  if (idx >= 256 * 64) return;
  int c = idx & 63, b = idx >> 6;
  float s = 0.f;
  for (int p = 0; p < 1024; ++p) s += (float)act[((size_t)b * 1024 + p) * 64 + c];
  feat[idx] = s * (1.f / 1024.f);
}

// ------------------------------- gate path ---------------------------------
__global__ __launch_bounds__(256) void k_gconv1(const float* __restrict__ x,
    const float* __restrict__ w, const float* __restrict__ bias,
    const float* __restrict__ g, const float* __restrict__ b,
    const float* __restrict__ m, const float* __restrict__ v,
    float* __restrict__ out) {
  int idx = blockIdx.x * 256 + threadIdx.x;           // 256*16*32*32
  int xx = idx & 31, yy = (idx >> 5) & 31, c = (idx >> 10) & 15, bb = idx >> 14;
  float acc = bias[c];
  for (int ci = 0; ci < 3; ++ci)
    for (int ky = 0; ky < 3; ++ky) {
      int iy = yy + ky - 1; if ((unsigned)iy >= 32u) continue;
      for (int kx = 0; kx < 3; ++kx) {
        int ix = xx + kx - 1; if ((unsigned)ix >= 32u) continue;
        acc += x[((bb * 3 + ci) * 32 + iy) * 32 + ix] * w[((c * 3 + ci) * 3 + ky) * 3 + kx];
      }
    }
  float sc = g[c] * rsqrtf(v[c] + BN_EPS);
  out[idx] = fmaxf((acc - m[c]) * sc + b[c], 0.f);
}

__global__ __launch_bounds__(256) void k_gpool(const float* __restrict__ in,
                                               float* __restrict__ out) {
  int idx = blockIdx.x * 256 + threadIdx.x;           // 256*16*16*16
  int xx = idx & 15, yy = (idx >> 4) & 15, c = (idx >> 8) & 15, bb = idx >> 12;
  float mx = -3.4e38f;
  for (int dy = 0; dy < 2; ++dy)
    for (int dx = 0; dx < 2; ++dx)
      mx = fmaxf(mx, in[((bb * 16 + c) * 32 + yy * 2 + dy) * 32 + xx * 2 + dx]);
  out[idx] = mx;
}

__global__ __launch_bounds__(256) void k_gconv2(const float* __restrict__ in,
    const float* __restrict__ w, const float* __restrict__ bias,
    const float* __restrict__ g, const float* __restrict__ b,
    const float* __restrict__ m, const float* __restrict__ v,
    float* __restrict__ out) {
  int idx = blockIdx.x * 256 + threadIdx.x;           // 256*32*16*16
  int xx = idx & 15, yy = (idx >> 4) & 15, c = (idx >> 8) & 31, bb = idx >> 13;
  float acc = bias[c];
  for (int ci = 0; ci < 16; ++ci)
    for (int ky = 0; ky < 3; ++ky) {
      int iy = yy + ky - 1; if ((unsigned)iy >= 16u) continue;
      for (int kx = 0; kx < 3; ++kx) {
        int ix = xx + kx - 1; if ((unsigned)ix >= 16u) continue;
        acc += in[((bb * 16 + ci) * 16 + iy) * 16 + ix] * w[((c * 16 + ci) * 3 + ky) * 3 + kx];
      }
    }
  float sc = g[c] * rsqrtf(v[c] + BN_EPS);
  out[idx] = fmaxf((acc - m[c]) * sc + b[c], 0.f);
}

__global__ __launch_bounds__(256) void k_gmean(const float* __restrict__ in,
                                               float* __restrict__ out) {
  int idx = blockIdx.x * 256 + threadIdx.x;           // 256*32
  if (idx >= 256 * 32) return;
  float s = 0.f;
  for (int p = 0; p < 256; ++p) s += in[idx * 256 + p];
  out[idx] = s * (1.f / 256.f);
}

__global__ __launch_bounds__(256) void k_ghead(const float* __restrict__ gm,
    const float* __restrict__ f1w, const float* __restrict__ f1b,
    const float* __restrict__ f2w, const float* __restrict__ f2b,
    float* __restrict__ gw) {
  int b = blockIdx.x * 256 + threadIdx.x;
  if (b >= 256) return;
  float h[64];
  for (int j = 0; j < 64; ++j) {
    float s = f1b[j];
    for (int i = 0; i < 32; ++i) s += gm[b * 32 + i] * f1w[j * 32 + i];
    h[j] = fmaxf(s, 0.f);
  }
  float p[4]; float mx = -3.4e38f;
  for (int e = 0; e < 4; ++e) {
    float s = f2b[e];
    for (int j = 0; j < 64; ++j) s += h[j] * f2w[e * 64 + j];
    p[e] = s; mx = fmaxf(mx, s);
  }
  float sum = 0.f;
  for (int e = 0; e < 4; ++e) { p[e] = expf(p[e] - mx); sum += p[e]; }
  for (int e = 0; e < 4; ++e) p[e] /= sum;
  int i1 = 0;
  for (int e = 1; e < 4; ++e) if (p[e] > p[i1]) i1 = e;       // first-index tie
  int i2 = -1;
  for (int e = 0; e < 4; ++e) { if (e == i1) continue; if (i2 < 0 || p[e] > p[i2]) i2 = e; }
  float wsum = p[i1] + p[i2];
  for (int e = 0; e < 4; ++e)
    gw[b * 4 + e] = (e == i1 || e == i2) ? p[e] / wsum : 0.f;
}

// ------------------------------- skip path ---------------------------------
__global__ __launch_bounds__(256) void k_sconv1(const float* __restrict__ x,
    const float* __restrict__ w, const float* __restrict__ bias,
    const float* __restrict__ g, const float* __restrict__ b,
    const float* __restrict__ m, const float* __restrict__ v,
    float* __restrict__ out) {
  int idx = blockIdx.x * 256 + threadIdx.x;           // 256*64*16*16
  int xx = idx & 15, yy = (idx >> 4) & 15, c = (idx >> 8) & 63, bb = idx >> 14;
  float acc = bias[c];
  for (int ci = 0; ci < 3; ++ci)
    for (int ky = 0; ky < 7; ++ky) {
      int iy = yy * 2 + ky - 3; if ((unsigned)iy >= 32u) continue;
      for (int kx = 0; kx < 7; ++kx) {
        int ix = xx * 2 + kx - 3; if ((unsigned)ix >= 32u) continue;
        acc += x[((bb * 3 + ci) * 32 + iy) * 32 + ix] * w[((c * 3 + ci) * 7 + ky) * 7 + kx];
      }
    }
  float sc = g[c] * rsqrtf(v[c] + BN_EPS);
  out[idx] = fmaxf((acc - m[c]) * sc + b[c], 0.f);
}

__global__ __launch_bounds__(256) void k_spool(const float* __restrict__ in,
                                               float* __restrict__ out) {
  int idx = blockIdx.x * 256 + threadIdx.x;           // 256*64*8*8
  int xx = idx & 7, yy = (idx >> 3) & 7, c = (idx >> 6) & 63, bb = idx >> 12;
  float mx = -3.4e38f;
  for (int ky = 0; ky < 3; ++ky) {
    int iy = yy * 2 + ky - 1; if ((unsigned)iy >= 16u) continue;
    for (int kx = 0; kx < 3; ++kx) {
      int ix = xx * 2 + kx - 1; if ((unsigned)ix >= 16u) continue;
      mx = fmaxf(mx, in[((bb * 64 + c) * 16 + iy) * 16 + ix]);
    }
  }
  out[idx] = mx;
}

__global__ __launch_bounds__(256) void k_sconv2(const float* __restrict__ in,
    const float* __restrict__ w, const float* __restrict__ bias,
    const float* __restrict__ g, const float* __restrict__ b,
    const float* __restrict__ m, const float* __restrict__ v,
    float* __restrict__ out) {
  int idx = blockIdx.x * 256 + threadIdx.x;           // 256*128*4*4
  int xx = idx & 3, yy = (idx >> 2) & 3, c = (idx >> 4) & 127, bb = idx >> 11;
  float acc = bias[c];
  for (int ci = 0; ci < 64; ++ci)
    for (int ky = 0; ky < 3; ++ky) {
      int iy = yy * 2 + ky - 1; if ((unsigned)iy >= 8u) continue;
      for (int kx = 0; kx < 3; ++kx) {
        int ix = xx * 2 + kx - 1; if ((unsigned)ix >= 8u) continue;
        acc += in[((bb * 64 + ci) * 8 + iy) * 8 + ix] * w[((c * 64 + ci) * 3 + ky) * 3 + kx];
      }
    }
  float sc = g[c] * rsqrtf(v[c] + BN_EPS);
  out[idx] = fmaxf((acc - m[c]) * sc + b[c], 0.f);
}

__global__ __launch_bounds__(256) void k_sconv3(const float* __restrict__ in,
    const float* __restrict__ w, const float* __restrict__ bias,
    const float* __restrict__ g, const float* __restrict__ b,
    const float* __restrict__ m, const float* __restrict__ v,
    float* __restrict__ out) {
  int idx = blockIdx.x * 256 + threadIdx.x;           // 256*256*2*2
  int xx = idx & 1, yy = (idx >> 1) & 1, c = (idx >> 2) & 255, bb = idx >> 10;
  float acc = bias[c];
  for (int ci = 0; ci < 128; ++ci)
    for (int ky = 0; ky < 3; ++ky) {
      int iy = yy * 2 + ky - 1; if ((unsigned)iy >= 4u) continue;
      for (int kx = 0; kx < 3; ++kx) {
        int ix = xx * 2 + kx - 1; if ((unsigned)ix >= 4u) continue;
        acc += in[((bb * 128 + ci) * 4 + iy) * 4 + ix] * w[((c * 128 + ci) * 3 + ky) * 3 + kx];
      }
    }
  float sc = g[c] * rsqrtf(v[c] + BN_EPS);
  out[idx] = fmaxf((acc - m[c]) * sc + b[c], 0.f);
}

__global__ __launch_bounds__(256) void k_smean(const float* __restrict__ in,
                                               float* __restrict__ out) {
  int idx = blockIdx.x * 256 + threadIdx.x;           // 256*256
  if (idx >= 256 * 256) return;
  float s = 0.f;
  for (int p = 0; p < 4; ++p) s += in[idx * 4 + p];
  out[idx] = s * 0.25f;
}

__global__ __launch_bounds__(256) void k_sfc(const float* __restrict__ in,
    const float* __restrict__ fw, const float* __restrict__ fb,
    float* __restrict__ out) {
  int idx = blockIdx.x * 256 + threadIdx.x;           // 256*64
  if (idx >= 256 * 64) return;
  int j = idx & 63, bb = idx >> 6;
  float s = fb[j];
  for (int i = 0; i < 256; ++i) s += in[bb * 256 + i] * fw[j * 256 + i];
  out[idx] = s;
}

// -------------------------- final combine + head ---------------------------
__global__ __launch_bounds__(256) void k_final(const float* __restrict__ gw,
    const float* __restrict__ efeat, const float* __restrict__ sfeat,
    const float* __restrict__ fw, const float* __restrict__ fb,
    float* __restrict__ out) {
  int idx = blockIdx.x * 256 + threadIdx.x;           // 256*10
  if (idx >= 256 * 10) return;
  int b = idx / 10, k = idx % 10;
  float acc = fb[k];
  for (int d = 0; d < 64; ++d) {
    float ff = 0.f;
    for (int e = 0; e < 4; ++e) ff += gw[b * 4 + e] * efeat[((size_t)e * 256 + b) * 64 + d];
    acc += ff * fw[k * 128 + d];
  }
  for (int d = 0; d < 64; ++d) acc += sfeat[b * 64 + d] * fw[k * 128 + 64 + d];
  out[idx] = acc;
}

// ---------------------------------------------------------------------------
extern "C" void kernel_launch(void* const* d_in, const int* in_sizes, int n_in,
                              void* d_out, int out_size, void* d_ws, size_t ws_size,
                              hipStream_t stream) {
  (void)in_sizes; (void)n_in; (void)out_size; (void)ws_size;
  // Canonical JAX pytree flatten order (sorted dict keys; tuples in order)
  const float* ex_bbn1_g = (const float*)d_in[0];
  const float* ex_bbn1_b = (const float*)d_in[1];
  const float* ex_bbn1_m = (const float*)d_in[2];
  const float* ex_bbn1_v = (const float*)d_in[3];
  const float* ex_bbn2_g = (const float*)d_in[4];
  const float* ex_bbn2_b = (const float*)d_in[5];
  const float* ex_bbn2_m = (const float*)d_in[6];
  const float* ex_bbn2_v = (const float*)d_in[7];
  const float* ex_bc1w   = (const float*)d_in[8];
  const float* ex_bc2w   = (const float*)d_in[9];
  const float* ex_bn1_g  = (const float*)d_in[10];
  const float* ex_bn1_b  = (const float*)d_in[11];
  const float* ex_bn1_m  = (const float*)d_in[12];
  const float* ex_bn1_v  = (const float*)d_in[13];
  const float* ex_c1w    = (const float*)d_in[14];
  // d_in[15]=experts.fcb, d_in[16]=experts.fcw  (dead: logits unused)
  const float* final_b   = (const float*)d_in[17];
  const float* final_w   = (const float*)d_in[18];
  const float* g_bn1_g = (const float*)d_in[19];
  const float* g_bn1_b = (const float*)d_in[20];
  const float* g_bn1_m = (const float*)d_in[21];
  const float* g_bn1_v = (const float*)d_in[22];
  const float* g_bn2_g = (const float*)d_in[23];
  const float* g_bn2_b = (const float*)d_in[24];
  const float* g_bn2_m = (const float*)d_in[25];
  const float* g_bn2_v = (const float*)d_in[26];
  const float* g_c1b = (const float*)d_in[27];
  const float* g_c1w = (const float*)d_in[28];
  const float* g_c2b = (const float*)d_in[29];
  const float* g_c2w = (const float*)d_in[30];
  const float* g_f1b = (const float*)d_in[31];
  const float* g_f1w = (const float*)d_in[32];
  const float* g_f2b = (const float*)d_in[33];
  const float* g_f2w = (const float*)d_in[34];
  const float* s_bn1_g = (const float*)d_in[35];
  const float* s_bn1_b = (const float*)d_in[36];
  const float* s_bn1_m = (const float*)d_in[37];
  const float* s_bn1_v = (const float*)d_in[38];
  const float* s_bn2_g = (const float*)d_in[39];
  const float* s_bn2_b = (const float*)d_in[40];
  const float* s_bn2_m = (const float*)d_in[41];
  const float* s_bn2_v = (const float*)d_in[42];
  const float* s_bn3_g = (const float*)d_in[43];
  const float* s_bn3_b = (const float*)d_in[44];
  const float* s_bn3_m = (const float*)d_in[45];
  const float* s_bn3_v = (const float*)d_in[46];
  const float* s_c1b = (const float*)d_in[47];
  const float* s_c1w = (const float*)d_in[48];
  const float* s_c2b = (const float*)d_in[49];
  const float* s_c2w = (const float*)d_in[50];
  const float* s_c3b = (const float*)d_in[51];
  const float* s_c3w = (const float*)d_in[52];
  const float* s_fb  = (const float*)d_in[53];
  const float* s_fw  = (const float*)d_in[54];
  const float* x     = (const float*)d_in[55];
  float* out = (float*)d_out;

  char* ws = (char*)d_ws;
  size_t off = 0;
  auto alloc = [&](size_t bytes) -> void* {
    void* p = ws + off; off += (bytes + 255) & ~(size_t)255; return p;
  };
  _Float16* actP  = (_Float16*)alloc((size_t)256 * 32 * 32 * 64 * 2);
  _Float16* actQ  = (_Float16*)alloc((size_t)256 * 32 * 32 * 64 * 2);
  _Float16* wt1   = (_Float16*)alloc((size_t)4 * 9 * 64 * 64 * 2);
  _Float16* wt2   = (_Float16*)alloc((size_t)4 * 9 * 64 * 64 * 2);
  float* efeat = (float*)alloc((size_t)4 * 256 * 64 * 4);
  float* g1    = (float*)alloc((size_t)256 * 16 * 32 * 32 * 4);
  float* g2    = (float*)alloc((size_t)256 * 16 * 16 * 16 * 4);
  float* g3    = (float*)alloc((size_t)256 * 32 * 16 * 16 * 4);
  float* gmean = (float*)alloc((size_t)256 * 32 * 4);
  float* gwt   = (float*)alloc((size_t)256 * 4 * 4);
  float* smean = (float*)alloc((size_t)256 * 256 * 4);
  float* sfeat = (float*)alloc((size_t)256 * 64 * 4);
  // Skip path reuses gate buffers (gate path finishes first on this stream)
  float* s1  = g1;                       // [256][64][16][16]
  float* s1p = g2;                       // [256][64][8][8]
  float* s2  = g3;                       // [256][128][4][4]
  float* s3  = s2 + (size_t)256 * 128 * 4 * 4;  // [256][256][2][2]

  // Repack expert block weights for WMMA B-fragments
  k_wcvt<<<576, 256, 0, stream>>>(ex_bc1w, wt1);
  k_wcvt<<<576, 256, 0, stream>>>(ex_bc2w, wt2);

  // Gate path
  k_gconv1<<<16384, 256, 0, stream>>>(x, g_c1w, g_c1b, g_bn1_g, g_bn1_b, g_bn1_m, g_bn1_v, g1);
  k_gpool<<<4096, 256, 0, stream>>>(g1, g2);
  k_gconv2<<<8192, 256, 0, stream>>>(g2, g_c2w, g_c2b, g_bn2_g, g_bn2_b, g_bn2_m, g_bn2_v, g3);
  k_gmean<<<32, 256, 0, stream>>>(g3, gmean);
  k_ghead<<<1, 256, 0, stream>>>(gmean, g_f1w, g_f1b, g_f2w, g_f2b, gwt);

  // Skip path
  k_sconv1<<<16384, 256, 0, stream>>>(x, s_c1w, s_c1b, s_bn1_g, s_bn1_b, s_bn1_m, s_bn1_v, s1);
  k_spool<<<4096, 256, 0, stream>>>(s1, s1p);
  k_sconv2<<<2048, 256, 0, stream>>>(s1p, s_c2w, s_c2b, s_bn2_g, s_bn2_b, s_bn2_m, s_bn2_v, s2);
  k_sconv3<<<1024, 256, 0, stream>>>(s2, s_c3w, s_c3b, s_bn3_g, s_bn3_b, s_bn3_m, s_bn3_v, s3);
  k_smean<<<256, 256, 0, stream>>>(s3, smean);
  k_sfc<<<64, 256, 0, stream>>>(smean, s_fw, s_fb, sfeat);

  // Experts (sequential; weight-shared residual block applied twice)
  const dim3 cgrid(256, 8);
  for (int e = 0; e < 4; ++e) {
    const size_t wo = (size_t)e * 9 * 64 * 64;
    k_econv1<<<65536, 256, 0, stream>>>(x, ex_c1w + (size_t)e * 64 * 27,
        ex_bn1_g + e * 64, ex_bn1_b + e * 64, ex_bn1_m + e * 64, ex_bn1_v + e * 64, actP);
    // block 1
    k_conv64_wmma<<<cgrid, 256, 0, stream>>>(actP, wt1 + wo,
        ex_bbn1_g + e * 64, ex_bbn1_b + e * 64, ex_bbn1_m + e * 64, ex_bbn1_v + e * 64,
        nullptr, actQ);
    k_conv64_wmma<<<cgrid, 256, 0, stream>>>(actQ, wt2 + wo,
        ex_bbn2_g + e * 64, ex_bbn2_b + e * 64, ex_bbn2_m + e * 64, ex_bbn2_v + e * 64,
        actP, actP);
    // block 2 (same weights)
    k_conv64_wmma<<<cgrid, 256, 0, stream>>>(actP, wt1 + wo,
        ex_bbn1_g + e * 64, ex_bbn1_b + e * 64, ex_bbn1_m + e * 64, ex_bbn1_v + e * 64,
        nullptr, actQ);
    k_conv64_wmma<<<cgrid, 256, 0, stream>>>(actQ, wt2 + wo,
        ex_bbn2_g + e * 64, ex_bbn2_b + e * 64, ex_bbn2_m + e * 64, ex_bbn2_v + e * 64,
        actP, actP);
    k_efeat<<<64, 256, 0, stream>>>(actP, efeat + (size_t)e * 256 * 64);
  }

  k_final<<<10, 256, 0, stream>>>(gwt, efeat, sfeat, final_w, final_b, out);
}
